// WindowAttention3D_65678639891289
// MI455X (gfx1250) — matrix operand order repaired and checked by reading, not compiled, not run on hardware
//
#include <hip/hip_runtime.h>
#include <hip/hip_bf16.h>

typedef __attribute__((ext_vector_type(16))) _Float16 v16h;
typedef __attribute__((ext_vector_type(8)))  _Float16 v8h;
typedef __attribute__((ext_vector_type(4)))  _Float16 v4h;
typedef __attribute__((ext_vector_type(8)))  float    v8f;

// ---- problem constants ----
constexpr int SQ    = 144;    // tokens per window
constexpr int DIM   = 384;
constexpr int HEADS = 12;
constexpr int HD    = 32;     // head dim
constexpr int NW    = 512;    // windows
constexpr int NQKV  = 3 * DIM;      // 1152
constexpr int VPAD  = 160;    // K padded to multiple of 32 for P@V
constexpr int NSLICE = 32;    // MLEV*MLAT (bias constant along MLON)

// ---- workspace layout (bytes, all offsets 256-aligned) ----
constexpr size_t SZ_X     = (size_t)NW * SQ * DIM * 2;        // x in f16
constexpr size_t SZ_WQKV  = (size_t)NQKV * DIM * 2;
constexpr size_t SZ_WPROJ = (size_t)DIM * DIM * 2;
constexpr size_t SZ_QK    = (size_t)NW * HEADS * SQ * HD * 2; // q or k
constexpr size_t SZ_VT    = (size_t)NW * HEADS * HD * VPAD * 2;
constexpr size_t SZ_BIAS  = (size_t)NSLICE * HEADS * SQ * SQ * 4;
constexpr size_t OFF_X     = 0;
constexpr size_t OFF_WQKV  = OFF_X + SZ_X;
constexpr size_t OFF_WPROJ = OFF_WQKV + SZ_WQKV;
constexpr size_t OFF_Q     = OFF_WPROJ + SZ_WPROJ;
constexpr size_t OFF_K     = OFF_Q + SZ_QK;
constexpr size_t OFF_VT    = OFF_K + SZ_QK;
constexpr size_t OFF_BIAS  = OFF_VT + SZ_VT;
constexpr size_t OFF_ATT   = OFF_BIAS + SZ_BIAS;

// ---- WMMA fragment helpers (layouts per CDNA5 ISA 7.12.2) ----
// A 16x32 f16: lane(li,hs): row = rowbase+li; elems 0..7 = K k0+hs*8.., elems 8..15 = +16
__device__ inline v16h loadA(const _Float16* __restrict__ rowp, int k) {
  v8h lo = *(const v8h*)(rowp + k);
  v8h hi = *(const v8h*)(rowp + k + 16);
  return __builtin_shufflevector(lo, hi, 0,1,2,3,4,5,6,7,8,9,10,11,12,13,14,15);
}
// B 32x16 f16: lane(li,hs): col = colbase+li; elems 0..7 = K k0+hs*8.., elems 8..15 = +16
__device__ inline v16h loadB(const _Float16* __restrict__ colp, int k) {
  const int kt = (k & ~31) + ((k & 16) >> 1);
  v8h lo = *(const v8h*)(colp + kt);
  v8h hi = *(const v8h*)(colp + kt + 16);
  return __builtin_shufflevector(lo, hi, 0,1,2,3,4,5,6,7,8,9,10,11,12,13,14,15);
}
__device__ inline v8f wmma32(v16h a, v16h b, v8f c) {
  return __builtin_amdgcn_wmma_f32_16x16x32_f16(false, a, false, b, (short)0, c,
                                                false, false);
}
__device__ inline v8f zero8() {
  v8f z = {0.f,0.f,0.f,0.f,0.f,0.f,0.f,0.f};
  return z;
}

// ---- kernel 1: f32 -> f16 convert (n4 = count/4) ----
__global__ void cvt_f16_kernel(const float* __restrict__ in,
                               _Float16* __restrict__ out, int n4) {
  int i = blockIdx.x * blockDim.x + threadIdx.x;
  if (i < n4) {
    float4 v = ((const float4*)in)[i];
    v4h o = { (_Float16)v.x, (_Float16)v.y, (_Float16)v.z, (_Float16)v.w };
    ((v4h*)out)[i] = o;
  }
}

// ---- kernel 2: zero the K-pad columns [144,160) of vT ----
__global__ void vpad_zero_kernel(_Float16* __restrict__ vT) {
  int row = blockIdx.x * blockDim.x + threadIdx.x;   // NW*HEADS*HD rows
  if (row < NW * HEADS * HD) {
    float4 z = make_float4(0.f, 0.f, 0.f, 0.f);
    float4* p = (float4*)(vT + (size_t)row * VPAD + SQ);
    p[0] = z; p[1] = z;
  }
}

// ---- kernel 3: precompute bias[(slice*HEADS+h)][n][m] = table[rel[n,m],lev,lat,h] ----
__global__ void bias_gather_kernel(const float* __restrict__ table,
                                   const int* __restrict__ rel,
                                   float* __restrict__ bias) {
  int sh = blockIdx.y;                 // slice*HEADS + h, 384 total
  int s = sh / HEADS, h = sh % HEADS;
  int lev = s >> 3, lat = s & 7;
  int e = blockIdx.x * blockDim.x + threadIdx.x;
  if (e < SQ * SQ) {
    int r = rel[e];
    bias[(size_t)sh * SQ * SQ + e] = table[((r * 4 + lev) * 8 + lat) * HEADS + h];
  }
}

// ---- kernel 4: QKV GEMM: 32x64 tile per wave, scatter to q/k/vT f16 ----
__global__ void __launch_bounds__(128)
qkv_gemm_kernel(const _Float16* __restrict__ X, const _Float16* __restrict__ W,
                const float* __restrict__ qkvb,
                _Float16* __restrict__ qb, _Float16* __restrict__ kb,
                _Float16* __restrict__ vT) {
  const int lane = threadIdx.x & 31;
  const int li = lane & 15, hs = lane >> 4;
  const int flat = blockIdx.x * 4 + (threadIdx.x >> 5);
  const int NG = NQKV / 64;                 // 18 column groups
  const int mt = flat / NG, g = flat % NG;  // mt over 32-row tiles (2304)
  const _Float16* arow0 = X + (size_t)(mt * 32 + li) * DIM;
  const _Float16* arow1 = arow0 + (size_t)16 * DIM;
  const _Float16* b0 = W + (size_t)(g * 64 +  0 + li) * DIM;
  const _Float16* b1 = W + (size_t)(g * 64 + 16 + li) * DIM;
  const _Float16* b2 = W + (size_t)(g * 64 + 32 + li) * DIM;
  const _Float16* b3 = W + (size_t)(g * 64 + 48 + li) * DIM;
  const int ka = hs * 8, kbo = hs * 16;
  v8f c[8];
#pragma unroll
  for (int i = 0; i < 8; i++) c[i] = zero8();
#pragma unroll
  for (int k0 = 0; k0 < DIM; k0 += 32) {
    v16h a0 = loadA(arow0, k0 + ka);
    v16h a1 = loadA(arow1, k0 + ka);
    v16h B0 = loadB(b0, k0 + kbo);
    c[0] = wmma32(a0, B0, c[0]);  c[4] = wmma32(a1, B0, c[4]);
    v16h B1 = loadB(b1, k0 + kbo);
    c[1] = wmma32(a0, B1, c[1]);  c[5] = wmma32(a1, B1, c[5]);
    v16h B2 = loadB(b2, k0 + kbo);
    c[2] = wmma32(a0, B2, c[2]);  c[6] = wmma32(a1, B2, c[6]);
    v16h B3 = loadB(b3, k0 + kbo);
    c[3] = wmma32(a0, B3, c[3]);  c[7] = wmma32(a1, B3, c[7]);
  }
  const float scale = 0.17677669529663687f;  // 32^-0.5 folded into q
#pragma unroll
  for (int p = 0; p < 2; p++) {
#pragma unroll
    for (int t = 0; t < 4; t++) {
      const int n = g * 64 + t * 16 + li;
      const int which = n / DIM;               // 0=q,1=k,2=v (uniform per tile)
      const int head = (n % DIM) >> 5;         // uniform per tile
      const int hd = n & 31;
      const float bv = qkvb[n];
#pragma unroll
      for (int r = 0; r < 8; r++) {
        float val = c[p * 4 + t][r] + bv;
        const int m = mt * 32 + p * 16 + r + hs * 8;
        const int w = m / SQ, tok = m % SQ;
        if (which == 0)
          qb[(((w * HEADS + head) * SQ + tok) << 5) + hd] = (_Float16)(val * scale);
        else if (which == 1)
          kb[(((w * HEADS + head) * SQ + tok) << 5) + hd] = (_Float16)val;
        else
          vT[(size_t)((w * HEADS + head) * HD + hd) * VPAD + tok] = (_Float16)val;
      }
    }
  }
}

// ---- kernel 5: attention, 3 waves/block, 1 wave per (window,head,16-row band) ----
__global__ void __launch_bounds__(96)
attn_kernel(const _Float16* __restrict__ qb, const _Float16* __restrict__ kb,
            const _Float16* __restrict__ vT, const float* __restrict__ bias,
            _Float16* __restrict__ att) {
  __shared__ _Float16 Pbuf[3][16 * VPAD];
  const int lane = threadIdx.x & 31;
  const int wid = threadIdx.x >> 5;           // 0..2
  const int li = lane & 15, hs = lane >> 4;
  const int band = (blockIdx.x % 3) * 3 + wid;  // 0..8
  const int wh = blockIdx.x / 3;
  const int w = wh / HEADS, h = wh % HEADS;
  _Float16* P = &Pbuf[wid][0];
  const _Float16* qp = qb + (size_t)wh * SQ * HD;
  const _Float16* kp = kb + (size_t)wh * SQ * HD;
  const _Float16* vp = vT + (size_t)wh * HD * VPAD;
  const float* bp = bias + (size_t)((w >> 4) * HEADS + h) * SQ * SQ;

  // S = (q*scale) @ k^T for this 16-row band: K = 32 -> one WMMA per 16x16 tile
  v16h aq = loadA(qp + (size_t)(band * 16 + li) * HD, hs * 8);
  v8f acc[9];
#pragma unroll
  for (int j = 0; j < 9; j++) acc[j] = zero8();
#pragma unroll
  for (int j = 0; j < 9; j++) {
    v16h bk = loadB(kp + (size_t)(j * 16 + li) * HD, hs * 16);
    acc[j] = wmma32(aq, bk, acc[j]);
  }

  // bias add + row-wise softmax; probs -> LDS as f16 A-operand source
#pragma unroll
  for (int r = 0; r < 8; r++) {
    const int row = band * 16 + r + hs * 8;    // global query row of this lane-half
    float s[9];
    float mx = -3.0e38f;
#pragma unroll
    for (int j = 0; j < 9; j++) {
      s[j] = acc[j][r] + bp[row * SQ + j * 16 + li];
      mx = fmaxf(mx, s[j]);
    }
    mx = fmaxf(mx, __shfl_xor(mx, 1, 32));
    mx = fmaxf(mx, __shfl_xor(mx, 2, 32));
    mx = fmaxf(mx, __shfl_xor(mx, 4, 32));
    mx = fmaxf(mx, __shfl_xor(mx, 8, 32));
    float sum = 0.f;
#pragma unroll
    for (int j = 0; j < 9; j++) { s[j] = __expf(s[j] - mx); sum += s[j]; }
    sum += __shfl_xor(sum, 1, 32);
    sum += __shfl_xor(sum, 2, 32);
    sum += __shfl_xor(sum, 4, 32);
    sum += __shfl_xor(sum, 8, 32);
    const float inv = 1.0f / sum;
#pragma unroll
    for (int j = 0; j < 9; j++)
      P[(r + hs * 8) * VPAD + j * 16 + li] = (_Float16)(s[j] * inv);
  }
  // zero-pad P cols [144,160): one 16B store per lane covers 16 rows x 16 cols
  {
    float4* z = (float4*)(P + (lane >> 1) * VPAD + SQ + (lane & 1) * 8);
    *z = make_float4(0.f, 0.f, 0.f, 0.f);
  }
  asm volatile("s_wait_dscnt 0" ::: "memory");   // wave-private LDS panel: no barrier needed

  // O = P @ V : K = 160 (5 steps), two 16x16 output tiles (hd 0..15, 16..31)
  v8f o0 = zero8(), o1 = zero8();
#pragma unroll
  for (int kk = 0; kk < 5; kk++) {
    const int k0 = kk * 32;
    v16h a   = loadA(P + li * VPAD, k0 + hs * 8);
    v16h bv0 = loadB(vp + (size_t)(li) * VPAD,      k0 + hs * 16);
    v16h bv1 = loadB(vp + (size_t)(16 + li) * VPAD, k0 + hs * 16);
    o0 = wmma32(a, bv0, o0);
    o1 = wmma32(a, bv1, o1);
  }
#pragma unroll
  for (int r = 0; r < 8; r++) {
    const int tok = band * 16 + r + hs * 8;
    _Float16* dst = att + (size_t)((w * SQ + tok) * HEADS + h) * HD;
    dst[li]      = (_Float16)o0[r];
    dst[16 + li] = (_Float16)o1[r];
  }
}

// ---- kernel 6: output proj: 32x64 tile per wave, f32 out + bias ----
__global__ void __launch_bounds__(128)
proj_gemm_kernel(const _Float16* __restrict__ A, const _Float16* __restrict__ W,
                 const float* __restrict__ pb, float* __restrict__ out) {
  const int lane = threadIdx.x & 31;
  const int li = lane & 15, hs = lane >> 4;
  const int flat = blockIdx.x * 4 + (threadIdx.x >> 5);
  const int NG = DIM / 64;                  // 6
  const int mt = flat / NG, g = flat % NG;  // mt over 32-row tiles (2304)
  const _Float16* arow0 = A + (size_t)(mt * 32 + li) * DIM;
  const _Float16* arow1 = arow0 + (size_t)16 * DIM;
  const _Float16* b0 = W + (size_t)(g * 64 +  0 + li) * DIM;
  const _Float16* b1 = W + (size_t)(g * 64 + 16 + li) * DIM;
  const _Float16* b2 = W + (size_t)(g * 64 + 32 + li) * DIM;
  const _Float16* b3 = W + (size_t)(g * 64 + 48 + li) * DIM;
  const int ka = hs * 8, kbo = hs * 16;
  v8f c[8];
#pragma unroll
  for (int i = 0; i < 8; i++) c[i] = zero8();
#pragma unroll
  for (int k0 = 0; k0 < DIM; k0 += 32) {
    v16h a0 = loadA(arow0, k0 + ka);
    v16h a1 = loadA(arow1, k0 + ka);
    v16h B0 = loadB(b0, k0 + kbo);
    c[0] = wmma32(a0, B0, c[0]);  c[4] = wmma32(a1, B0, c[4]);
    v16h B1 = loadB(b1, k0 + kbo);
    c[1] = wmma32(a0, B1, c[1]);  c[5] = wmma32(a1, B1, c[5]);
    v16h B2 = loadB(b2, k0 + kbo);
    c[2] = wmma32(a0, B2, c[2]);  c[6] = wmma32(a1, B2, c[6]);
    v16h B3 = loadB(b3, k0 + kbo);
    c[3] = wmma32(a0, B3, c[3]);  c[7] = wmma32(a1, B3, c[7]);
  }
#pragma unroll
  for (int p = 0; p < 2; p++) {
#pragma unroll
    for (int t = 0; t < 4; t++) {
      const int n = g * 64 + t * 16 + li;
      const float bv = pb[n];
#pragma unroll
      for (int r = 0; r < 8; r++) {
        const int m = mt * 32 + p * 16 + r + hs * 8;
        out[(size_t)m * DIM + n] = c[p * 4 + t][r] + bv;
      }
    }
  }
}

extern "C" void kernel_launch(void* const* d_in, const int* in_sizes, int n_in,
                              void* d_out, int out_size, void* d_ws, size_t ws_size,
                              hipStream_t stream) {
  (void)in_sizes; (void)n_in; (void)out_size; (void)ws_size;
  const float* x      = (const float*)d_in[0];
  const float* qkv_w  = (const float*)d_in[1];
  const float* qkv_b  = (const float*)d_in[2];
  const float* proj_w = (const float*)d_in[3];
  const float* proj_b = (const float*)d_in[4];
  const float* table  = (const float*)d_in[5];
  const int*   rel    = (const int*)d_in[6];
  float* out = (float*)d_out;

  char* ws = (char*)d_ws;
  _Float16* xf    = (_Float16*)(ws + OFF_X);
  _Float16* wqkv  = (_Float16*)(ws + OFF_WQKV);
  _Float16* wproj = (_Float16*)(ws + OFF_WPROJ);
  _Float16* qbuf  = (_Float16*)(ws + OFF_Q);
  _Float16* kbuf  = (_Float16*)(ws + OFF_K);
  _Float16* vTb   = (_Float16*)(ws + OFF_VT);
  float*    biasb = (float*)(ws + OFF_BIAS);
  _Float16* attb  = (_Float16*)(ws + OFF_ATT);

  // converts
  {
    int n4 = NW * SQ * DIM / 4;
    cvt_f16_kernel<<<(n4 + 255) / 256, 256, 0, stream>>>(x, xf, n4);
  }
  {
    int n4 = NQKV * DIM / 4;
    cvt_f16_kernel<<<(n4 + 255) / 256, 256, 0, stream>>>(qkv_w, wqkv, n4);
  }
  {
    int n4 = DIM * DIM / 4;
    cvt_f16_kernel<<<(n4 + 255) / 256, 256, 0, stream>>>(proj_w, wproj, n4);
  }
  // zero pad of vT K-columns
  vpad_zero_kernel<<<(NW * HEADS * HD + 255) / 256, 256, 0, stream>>>(vTb);
  // bias precompute gather
  bias_gather_kernel<<<dim3((SQ * SQ + 255) / 256, NSLICE * HEADS), 256, 0, stream>>>(
      table, rel, biasb);
  // QKV projection: (NW*SQ/32) m-tiles * 18 n-groups waves, 4 waves/block
  {
    int waves = (NW * SQ / 32) * (NQKV / 64);   // 41472
    qkv_gemm_kernel<<<waves / 4, 128, 0, stream>>>(xf, wqkv, qkv_b, qbuf, kbuf, vTb);
  }
  // attention: one wave per (window, head, band); 3 band-waves per block
  attn_kernel<<<NW * HEADS * 3, 96, 0, stream>>>(qbuf, kbuf, vTb, biasb, attb);
  // output projection
  {
    int waves = (NW * SQ / 32) * (DIM / 64);    // 13824
    proj_gemm_kernel<<<waves / 4, 128, 0, stream>>>(attb, wproj, proj_b, out);
  }
}
